// MultiHeadAttention_50440095924383
// MI455X (gfx1250) — matrix operand order repaired and checked by reading, not compiled, not run on hardware
//
#include <hip/hip_runtime.h>
#include <hip/hip_bf16.h>

// MI455X / gfx1250 multi-head attention forward.
// B=2, S=2048, E=1024, H=16, HD=64. All GEMMs via v_wmma_f32_16x16x32_bf16.
//
// Pipeline (all in d_ws, ~48MB):
//   1) convert x, W_q, W_k, W_v, W_o  fp32 -> bf16
//   2) Q/K/V = x @ W^T   (WMMA gemm, K=64 pipeline stages double-buffered in
//                         LDS via GLOBAL_LOAD_ASYNC_TO_LDS_B128 + s_wait_asynccnt)
//   3) flash attention per (b,h,64-row q block)  (WMMA + LDS + online softmax)
//   4) out = AO @ W_o^T  (same gemm, fp32 out)

typedef __bf16 bf16;
typedef bf16  v16bf __attribute__((ext_vector_type(16)));
typedef bf16  v8bf  __attribute__((ext_vector_type(8)));
typedef float v8f   __attribute__((ext_vector_type(8)));

#define EMBED 1024
#define SEQ   2048
#define NB    2
#define NH    16
#define HD    64
#define MROWS (NB * SEQ)   // 4096

// ---------------------------------------------------------------------------
// WMMA fragment loaders (CDNA5 ISA 7.12.2 VGPR layouts, wave32).
//
// A-matrix 16x32 bf16: lane holds row M = lane&15.
//   lanes 0-15 : K = {0..7, 16..23};  lanes 16-31: K = {8..15, 24..31}
//   -> per lane: two contiguous 8-elem (16B) chunks at kb and kb+16, kb=(lane>>4)*8.
// B-matrix 32x16 bf16: lane holds col N = lane&15.
//   lanes 0-15 : K = {0..7, 16..23};  lanes 16-31: K = {8..15, 24..31}  (same as A)
// C/D 16x16 f32: VGPR v, lane l -> M = v + 8*(l>>4), N = l&15.
// ---------------------------------------------------------------------------
template <typename PT>
static __device__ inline v16bf load_a_frag(const PT* base, int ldk, int lane) {
  const int r  = lane & 15;
  const int kb = (lane >> 4) * 8;
  v8bf lo = *(const v8bf*)(base + (size_t)r * ldk + kb);
  v8bf hi = *(const v8bf*)(base + (size_t)r * ldk + kb + 16);
  v16bf a;
#pragma unroll
  for (int i = 0; i < 8; ++i) { a[i] = lo[i]; a[i + 8] = hi[i]; }
  return a;
}

template <typename PT>
static __device__ inline v16bf load_b_frag(const PT* base, int ldk, int lane) {
  // base points at element (n=0, k=0) of a [16 x 32] tile stored n-major,
  // k contiguous with stride ldk (e.g. a row-major weight matrix W[N][K]).
  const int n  = lane & 15;
  const int kb = (lane >> 4) * 8;
  v8bf lo = *(const v8bf*)(base + (size_t)n * ldk + kb);
  v8bf hi = *(const v8bf*)(base + (size_t)n * ldk + kb + 16);
  v16bf b;
#pragma unroll
  for (int i = 0; i < 8; ++i) { b[i] = lo[i]; b[i + 8] = hi[i]; }
  return b;
}

static __device__ inline v8f wmma_bf16(v16bf a, v16bf b, v8f c) {
  return __builtin_amdgcn_wmma_f32_16x16x32_bf16(
      /*neg_a=*/false, a, /*neg_b=*/false, b,
      /*c_mod=*/(short)0, c, /*reuse_a=*/false, /*reuse_b=*/false);
}

// 16-byte global -> LDS async copy (ASYNCcnt-tracked, no VGPR data movement).
// LDS operand is the 32-bit group-segment offset (generic LDS addr truncates
// to the DS address per the flat-aperture rules).
static __device__ inline void async_cp16(bf16* lds_dst, const bf16* gsrc) {
  unsigned           l = (unsigned)(uintptr_t)lds_dst;
  unsigned long long g = (unsigned long long)(uintptr_t)gsrc;
  asm volatile("global_load_async_to_lds_b128 %0, %1, off"
               :: "v"(l), "v"(g) : "memory");
}

// ---------------------------------------------------------------------------
// fp32 -> bf16 conversion
// ---------------------------------------------------------------------------
__global__ __launch_bounds__(256) void cvt_f32_bf16(const float* __restrict__ in,
                                                    bf16* __restrict__ out, int n) {
  int i = blockIdx.x * blockDim.x + threadIdx.x;
  int stride = gridDim.x * blockDim.x;
  for (; i < n; i += stride) out[i] = (bf16)in[i];
}

// ---------------------------------------------------------------------------
// GEMM: out[M,N] = A[M,K] @ W[N,K]^T   (torch Linear, bias=False)
// Block = 256 threads = 8 waves arranged 2(M) x 4(N); wave tile 64x32,
// block tile 128x128. Pipeline stage depth K=64 (two 32-deep WMMA substeps).
// A/B stage tiles (128x64 bf16, 16 KB each) are double-buffered in LDS and
// filled with GLOBAL_LOAD_ASYNC_TO_LDS_B128 (8 asyncs/thread/stage); stage
// n+1's DMA overlaps stage n's 16 WMMAs, fenced by s_wait_asynccnt. 64 KB
// LDS total -> up to 5 workgroups per 320 KB WGP.
// ---------------------------------------------------------------------------
template <bool F32OUT>
__global__ __launch_bounds__(256) void gemm_xwT(const bf16* __restrict__ A,
                                                const bf16* __restrict__ W,
                                                void* __restrict__ out,
                                                int M, int N, int K) {
  __shared__ alignas(16) bf16 sA[2][128 * 64];   // [buf][row][k] : 2 x 16 KB
  __shared__ alignas(16) bf16 sB[2][128 * 64];   // [buf][n]  [k] : 2 x 16 KB

  const int tid  = threadIdx.x;
  const int lane = tid & 31;
  const int wave = tid >> 5;                  // 0..7
  const int wm = wave >> 2;                   // 0..1
  const int wn = wave & 3;                    // 0..3
  const int m0 = blockIdx.x * 128;
  const int n0 = blockIdx.y * 128;

  // Each thread stages 64B of the A tile and 64B of the B tile (8 x b128).
  const int srow = tid >> 1;                  // 0..127
  const int se0  = (tid & 1) * 32;            // element offset: 0 or 32

  auto stage = [&](int buf, int k0) {
    const bf16* ga = A + (size_t)(m0 + srow) * K + k0 + se0;
    const bf16* gw = W + (size_t)(n0 + srow) * K + k0 + se0;
    bf16* la = &sA[buf][srow * 64 + se0];
    bf16* lw = &sB[buf][srow * 64 + se0];
#pragma unroll
    for (int c8 = 0; c8 < 4; ++c8) {
      async_cp16(la + c8 * 8, ga + c8 * 8);
      async_cp16(lw + c8 * 8, gw + c8 * 8);
    }
  };

  v8f c[4][2];
#pragma unroll
  for (int i = 0; i < 4; ++i)
#pragma unroll
    for (int j = 0; j < 2; ++j) c[i][j] = (v8f){0.f, 0.f, 0.f, 0.f, 0.f, 0.f, 0.f, 0.f};

  stage(0, 0);

  for (int k0 = 0; k0 < K; k0 += 64) {
    const int buf = (k0 >> 6) & 1;
    if (k0 + 64 < K) {
      stage(buf ^ 1, k0 + 64);
      // 16 asyncs outstanding; wait until only the 8 just-issued remain,
      // i.e. buffer `buf` is fully written.
      asm volatile("s_wait_asynccnt 0x8" ::: "memory");
    } else {
      asm volatile("s_wait_asynccnt 0x0" ::: "memory");
    }
    __syncthreads();   // buffer `buf` visible to all waves

#pragma unroll
    for (int ks = 0; ks < 64; ks += 32) {
      v16bf b0 = load_b_frag(&sB[buf][(wn * 32 + 0)  * 64 + ks], 64, lane);
      v16bf b1 = load_b_frag(&sB[buf][(wn * 32 + 16) * 64 + ks], 64, lane);
#pragma unroll
      for (int i = 0; i < 4; ++i) {
        v16bf a = load_a_frag(&sA[buf][(wm * 64 + i * 16) * 64 + ks], 64, lane);
        c[i][0] = wmma_bf16(a, b0, c[i][0]);
        c[i][1] = wmma_bf16(a, b1, c[i][1]);
      }
    }

    __syncthreads();   // all waves done reading `buf` before it is re-staged
  }

  const int half = lane >> 4;
  const int col0 = lane & 15;
#pragma unroll
  for (int i = 0; i < 4; ++i)
#pragma unroll
    for (int j = 0; j < 2; ++j)
#pragma unroll
      for (int vr = 0; vr < 8; ++vr) {
        const int row = m0 + wm * 64 + i * 16 + vr + 8 * half;
        const int col = n0 + wn * 32 + j * 16 + col0;
        if (F32OUT)
          ((float*)out)[(size_t)row * N + col] = c[i][j][vr];
        else
          ((bf16*)out)[(size_t)row * N + col] = (bf16)c[i][j][vr];
      }
}

// ---------------------------------------------------------------------------
// Flash attention, causal. Grid: (S/64, B*H). Block: 128 threads = 4 waves.
// Each wave owns 16 query rows. K-blocks of 64 keys, online softmax.
//   scores: A = Q frag (regs), B = K rows (head-dim contiguous -> direct load)
//   PV:     A = P (restaged via LDS), B = V^T staged in LDS cooperatively
// (V staging needs a transpose, which the async-to-LDS path cannot do --
//  per-lane copies are contiguous-byte only -- so it stays VALU+ds_store.)
// ---------------------------------------------------------------------------
__global__ __launch_bounds__(128) void flash_attn(const bf16* __restrict__ Q,
                                                  const bf16* __restrict__ Km,
                                                  const bf16* __restrict__ Vm,
                                                  bf16* __restrict__ AO) {
  __shared__ alignas(16) bf16 sVt[HD * 64];       // Vt[dim][key]  : 8 KB
  __shared__ alignas(16) bf16 sP[4][16 * 64];     // per-wave P    : 8 KB

  const int lane = threadIdx.x & 31;
  const int wave = threadIdx.x >> 5;  // 0..3
  const int qblk = blockIdx.x;        // 0..31
  const int bh   = blockIdx.y;
  const int b    = bh >> 4;
  const int h    = bh & 15;
  const int half = lane >> 4;
  const int col  = lane & 15;

  // Q fragments for this wave's 16 rows (held in registers the whole time)
  const bf16* Qbase = Q + ((size_t)(b * SEQ + qblk * 64 + wave * 16)) * EMBED + h * HD;
  v16bf qf0 = load_a_frag(Qbase + 0,  EMBED, lane);
  v16bf qf1 = load_a_frag(Qbase + 32, EMBED, lane);

  v8f o[4];
#pragma unroll
  for (int j = 0; j < 4; ++j) o[j] = (v8f){0.f, 0.f, 0.f, 0.f, 0.f, 0.f, 0.f, 0.f};
  float mrow[8], lrow[8];
#pragma unroll
  for (int vr = 0; vr < 8; ++vr) { mrow[vr] = -1e30f; lrow[vr] = 0.f; }

  for (int kblk = 0; kblk <= qblk; ++kblk) {
    __syncthreads();  // previous iteration done reading sVt

    // --- stage V^T into LDS: 128 threads x 32 elems -------------------------
    {
      const int key = threadIdx.x >> 1;           // 0..63
      const int d0  = (threadIdx.x & 1) * 32;     // 0 or 32
      const bf16* vsrc =
          Vm + ((size_t)(b * SEQ + kblk * 64 + key)) * EMBED + h * HD + d0;
#pragma unroll
      for (int c8 = 0; c8 < 4; ++c8) {
        v8bf v = *(const v8bf*)(vsrc + c8 * 8);
#pragma unroll
        for (int e = 0; e < 8; ++e) sVt[(d0 + c8 * 8 + e) * 64 + key] = v[e];
      }
    }

    // --- scores S = Q K^T (overlaps the V staging) --------------------------
    v8f s[4];
#pragma unroll
    for (int j = 0; j < 4; ++j) {
      const bf16* Kbase =
          Km + ((size_t)(b * SEQ + kblk * 64 + j * 16)) * EMBED + h * HD;
      v8f acc = (v8f){0.f, 0.f, 0.f, 0.f, 0.f, 0.f, 0.f, 0.f};
      acc = wmma_bf16(qf0, load_b_frag(Kbase + 0,  EMBED, lane), acc);
      acc = wmma_bf16(qf1, load_b_frag(Kbase + 32, EMBED, lane), acc);
      s[j] = acc;
    }

    // --- scale + causal mask ------------------------------------------------
    const bool diag = (kblk == qblk);
#pragma unroll
    for (int j = 0; j < 4; ++j)
#pragma unroll
      for (int vr = 0; vr < 8; ++vr) {
        float v = s[j][vr] * 0.125f;  // 1/sqrt(64)
        if (diag) {
          const int kidx = j * 16 + col;
          const int qidx = wave * 16 + vr + 8 * half;
          if (kidx > qidx) v = -1e30f;
        }
        s[j][vr] = v;
      }

    // --- online softmax: row max (16-lane reduce), rescale, exp, row sum ----
    float alpha[8];
#pragma unroll
    for (int vr = 0; vr < 8; ++vr) {
      float m = s[0][vr];
#pragma unroll
      for (int j = 1; j < 4; ++j) m = fmaxf(m, s[j][vr]);
#pragma unroll
      for (int x = 8; x >= 1; x >>= 1) m = fmaxf(m, __shfl_xor(m, x, 32));
      const float mn = fmaxf(mrow[vr], m);
      alpha[vr] = __expf(mrow[vr] - mn);
      mrow[vr]  = mn;
    }

    float psum[8];
#pragma unroll
    for (int vr = 0; vr < 8; ++vr) psum[vr] = 0.f;
#pragma unroll
    for (int j = 0; j < 4; ++j)
#pragma unroll
      for (int vr = 0; vr < 8; ++vr) {
        const float p = __expf(s[j][vr] - mrow[vr]);
        psum[vr] += p;
        sP[wave][(vr + 8 * half) * 64 + j * 16 + col] = (bf16)p;
      }
#pragma unroll
    for (int vr = 0; vr < 8; ++vr) {
#pragma unroll
      for (int x = 8; x >= 1; x >>= 1) psum[vr] += __shfl_xor(psum[vr], x, 32);
      lrow[vr] = lrow[vr] * alpha[vr] + psum[vr];
#pragma unroll
      for (int j = 0; j < 4; ++j) o[j][vr] *= alpha[vr];
    }

    __syncthreads();  // sVt fully staged by all waves

    // --- O += P @ V ---------------------------------------------------------
#pragma unroll
    for (int t = 0; t < 2; ++t) {
      v16bf pf = load_a_frag(&sP[wave][0] + t * 32, 64, lane);
#pragma unroll
      for (int j = 0; j < 4; ++j) {
        v16bf vf = load_b_frag(&sVt[0] + (size_t)(j * 16) * 64 + t * 32, 64, lane);
        o[j] = wmma_bf16(pf, vf, o[j]);
      }
    }
  }

  // --- epilogue: normalize and store bf16 into AO ---------------------------
#pragma unroll
  for (int vr = 0; vr < 8; ++vr) {
    const float inv = 1.0f / lrow[vr];
    const int row = qblk * 64 + wave * 16 + vr + 8 * half;
#pragma unroll
    for (int j = 0; j < 4; ++j) {
      AO[((size_t)(b * SEQ + row)) * EMBED + h * HD + j * 16 + col] =
          (bf16)(o[j][vr] * inv);
    }
  }
}

// ---------------------------------------------------------------------------
// Host launcher
// ---------------------------------------------------------------------------
extern "C" void kernel_launch(void* const* d_in, const int* in_sizes, int n_in,
                              void* d_out, int out_size, void* d_ws, size_t ws_size,
                              hipStream_t stream) {
  (void)in_sizes; (void)n_in; (void)out_size; (void)ws_size;

  const float* x  = (const float*)d_in[0];
  // d_in[1] = mask (causal; handled analytically)
  const float* Wq = (const float*)d_in[2];
  const float* Wk = (const float*)d_in[3];
  const float* Wv = (const float*)d_in[4];
  const float* Wo = (const float*)d_in[5];
  float* out = (float*)d_out;

  char* ws = (char*)d_ws;
  const size_t MB = 1024 * 1024;
  bf16* xb  = (bf16*)(ws + 0 * MB);    // 4096x1024 bf16 : 8 MB
  bf16* Wqb = (bf16*)(ws + 8 * MB);    // 1024x1024 bf16 : 2 MB
  bf16* Wkb = (bf16*)(ws + 10 * MB);
  bf16* Wvb = (bf16*)(ws + 12 * MB);
  bf16* Wob = (bf16*)(ws + 14 * MB);
  bf16* Qb  = (bf16*)(ws + 16 * MB);   // 8 MB
  bf16* Kb  = (bf16*)(ws + 24 * MB);   // 8 MB
  bf16* Vb  = (bf16*)(ws + 32 * MB);   // 8 MB
  bf16* AOb = (bf16*)(ws + 40 * MB);   // 8 MB  (total 48 MB)

  const int nX = MROWS * EMBED;        // 4194304
  const int nW = EMBED * EMBED;        // 1048576
  cvt_f32_bf16<<<1024, 256, 0, stream>>>(x,  xb,  nX);
  cvt_f32_bf16<<<512,  256, 0, stream>>>(Wq, Wqb, nW);
  cvt_f32_bf16<<<512,  256, 0, stream>>>(Wk, Wkb, nW);
  cvt_f32_bf16<<<512,  256, 0, stream>>>(Wv, Wvb, nW);
  cvt_f32_bf16<<<512,  256, 0, stream>>>(Wo, Wob, nW);

  dim3 ggrid(MROWS / 128, EMBED / 128);  // 32 x 8
  gemm_xwT<false><<<ggrid, 256, 0, stream>>>(xb, Wqb, (void*)Qb, MROWS, EMBED, EMBED);
  gemm_xwT<false><<<ggrid, 256, 0, stream>>>(xb, Wkb, (void*)Kb, MROWS, EMBED, EMBED);
  gemm_xwT<false><<<ggrid, 256, 0, stream>>>(xb, Wvb, (void*)Vb, MROWS, EMBED, EMBED);

  dim3 agrid(SEQ / 64, NB * NH);         // 32 x 32
  flash_attn<<<agrid, 128, 0, stream>>>(Qb, Kb, Vb, AOb);

  gemm_xwT<true><<<ggrid, 256, 0, stream>>>(AOb, Wob, (void*)out, MROWS, EMBED, EMBED);
}
